// hier_node_vae_32684701123156
// MI455X (gfx1250) — matrix operand order repaired and checked by reading, not compiled, not run on hardware
//
#include <hip/hip_runtime.h>

// ---------------------------------------------------------------------------
// MI455X (gfx1250) fused hierarchical-VAE decoder.
// ~584 GFLOP vs ~0.4 GB essential HBM traffic -> strongly compute-bound:
// every matmul runs on v_wmma_f32_16x16x32_f16, whole layer chain fused so
// intermediates live in LDS. Register blocking: 2 M-tiles x 4 N-tiles per
// k-step (8 independent WMMA chains, A read once from LDS per k-step).
// ---------------------------------------------------------------------------

typedef __attribute__((ext_vector_type(16))) _Float16 v16h;
typedef __attribute__((ext_vector_type(8)))  _Float16 v8h;
typedef __attribute__((ext_vector_type(8)))  float    v8f;

union V16 { v16h v; v8h h[2]; };

// Weight arena offsets (in halfs). Each matrix stored as (NTpad x KT) tiles of
// 512 halfs, tile (kt,nt) at ((nt*KT)+kt)*512, pre-swizzled into the wave32
// WMMA B layout, contiguous k = 16*(lane/16)+e (as A loads). NTpad rounded to multiple of 4 so
// the NTB=4 register blocking divides evenly (padding tiles are zero-filled).
static constexpr size_t T_WLD = 0;        // 64x128   : 2x8
static constexpr size_t T_WRD = 8192;     // 128x768  : 4x48
static constexpr size_t T_WFD = 106496;   // 256x512  : 8x32
static constexpr size_t T_WC1 = 237568;   // 256x256  : 8x16
static constexpr size_t T_WC2 = 303104;
static constexpr size_t T_WC3 = 368640;
static constexpr size_t T_WCD = 434176;   // 256x27   : 8x4  (pad 2->4)
static constexpr size_t T_WCN = 450560;   // 128x256  : 4x16
static constexpr size_t T_WND = 483328;   // 256x512  : 8x32
static constexpr size_t T_WN1 = 614400;   // 512x512  : 16x32
static constexpr size_t T_WN2 = 876544;
static constexpr size_t T_WA0 = 1138688;  // 512x1024 : 16x64
static constexpr size_t T_WA1 = 1662976;  // 1024x512 : 32x32
static constexpr size_t T_WN3 = 2187264;  // 512x101  : 16x8 (pad 7->8)
static constexpr size_t T_END = 2252800;  // ~4.5 MB f16 (L2 resident)

#define AW 1024            // bufA width (halfs)
#define BW 512             // bufB width (halfs)
#define SMEM_BYTES ((32 * AW + 32 * BW) * 2)   // 96 KB -> 3 WGs / WGP

// --- f32 -> f16 weight pack into WMMA B-tile layout ------------------------
// dst[tile*512 + lane*16 + e] = W[k][n],
//   k = kt*32 + (lane<16 ? 0 : 16) + e,  n = nt*16 + (lane&15); OOB -> 0
__global__ void pack_weights(const float* __restrict__ W,
                             _Float16* __restrict__ dst,
                             int K, int N, int KT, int NTpad) {
  int idx = blockIdx.x * blockDim.x + threadIdx.x;
  int total = KT * NTpad * 512;
  if (idx >= total) return;
  int tile   = idx >> 9;
  int within = idx & 511;
  int lane   = within >> 4;
  int e      = within & 15;
  int nt = tile / KT;
  int kt = tile - nt * KT;
  int k = kt * 32 + ((lane >> 4) << 4) + e;
  int n = nt * 16 + (lane & 15);
  float v = (k < K && n < N) ? W[(size_t)k * N + n] : 0.0f;
  dst[idx] = (_Float16)v;
}

__device__ __forceinline__ void lds_fence() {
  asm volatile("s_wait_dscnt 0" ::: "memory");
}

// --- per-wave GEMM over a 32-row slab --------------------------------------
// MODE 0: relu -> LDS f16
// MODE 1: sigmoid -> global f32
// MODE 2: sigmoid(2x) (TEMP=0.5) -> global f32, column-masked
// MODE 3: relu + add(addBase) -> LDS f16   (nb_h + relu(ctn))
template <int MODE, int NTB>
__device__ __forceinline__ void wave_gemm(
    const _Float16* __restrict__ wt, const float* __restrict__ bias,
    const _Float16* aBase, int aW,
    _Float16* dBase, int dW,
    float* gBase, int gW,
    const _Float16* addBase, int addW,
    int Kin, int NTtiles, int Nout, int lane) {
  const int KT = Kin >> 5;
  const int m     = lane & 15;
  const int kb    = (lane < 16) ? 0 : 16;  // A K-base per half-wave (contiguous k order)
  const int rOff  = (lane < 16) ? 0 : 8;   // f32 C/D layout row offset
  const int nlane = lane & 15;
  const _Float16* aRow0 = aBase + m * aW;          // M-tile 0 (rows 0..15)
  const _Float16* aRow1 = aBase + (m + 16) * aW;   // M-tile 1 (rows 16..31)
  const _Float16* wl = wt + lane * 16;
#pragma unroll 1
  for (int nt0 = 0; nt0 < NTtiles; nt0 += NTB) {
    v8f acc0[NTB];
    v8f acc1[NTB];
#pragma unroll
    for (int j = 0; j < NTB; ++j) { acc0[j] = (v8f){}; acc1[j] = (v8f){}; }
    const _Float16* wg = wl + (size_t)nt0 * KT * 512;
#pragma unroll 1
    for (int kt = 0; kt < KT; ++kt) {
      const int ko = kt * 32 + kb;
      V16 a0, a1;                                   // A loaded ONCE per k-step
      a0.h[0] = *(const v8h*)(aRow0 + ko);          // K = ko..ko+7
      a0.h[1] = *(const v8h*)(aRow0 + ko + 8);      // K = ko+8..ko+15
      a1.h[0] = *(const v8h*)(aRow1 + ko);
      a1.h[1] = *(const v8h*)(aRow1 + ko + 8);
#pragma unroll
      for (int j = 0; j < NTB; ++j) {               // 4 B streams, 8 WMMAs
        v16h bm = *(const v16h*)(wg + ((size_t)j * KT + kt) * 512);
        __builtin_prefetch(wg + ((size_t)j * KT + kt + 2) * 512, 0, 3);
        acc0[j] = __builtin_amdgcn_wmma_f32_16x16x32_f16(
            false, a0.v, false, bm, (short)0, acc0[j], false, false);
        acc1[j] = __builtin_amdgcn_wmma_f32_16x16x32_f16(
            false, a1.v, false, bm, (short)0, acc1[j], false, false);
      }
    }
#pragma unroll
    for (int j = 0; j < NTB; ++j) {
      const int col = (nt0 + j) * 16 + nlane;
      const bool cok = col < Nout;
      const float bv = cok ? bias[col] : 0.0f;
#pragma unroll
      for (int i = 0; i < 8; ++i) {
        const int r0 = rOff + i;      // row in M-tile 0; +16 for M-tile 1
        float v0 = acc0[j][i] + bv;
        float v1 = acc1[j][i] + bv;
        if (MODE == 0 || MODE == 3) {
          v0 = fmaxf(v0, 0.0f);
          v1 = fmaxf(v1, 0.0f);
          if (MODE == 3) {
            v0 += (float)addBase[r0 * addW + col];
            v1 += (float)addBase[(r0 + 16) * addW + col];
          }
          dBase[r0 * dW + col]        = (_Float16)v0;
          dBase[(r0 + 16) * dW + col] = (_Float16)v1;
        } else {
          const float s = (MODE == 2) ? -2.0f : -1.0f;  // /TEMP fused into exp
          v0 = 1.0f / (1.0f + __expf(s * v0));
          v1 = 1.0f / (1.0f + __expf(s * v1));
          if (cok) {
            gBase[(size_t)r0 * gW + col]        = v0;
            gBase[((size_t)r0 + 16) * gW + col] = v1;
          }
        }
      }
    }
  }
}

struct Biases {
  const float *ld, *rd, *fd, *c1, *c2, *c3, *cd, *cn, *nd, *n1, *n2, *a0, *a1, *n3;
};

// --- fused decoder: one wave per 32-row slab, full layer chain in LDS ------
__global__ __launch_bounds__(32) void fused_decoder(
    const float* __restrict__ z, const int* __restrict__ cind,
    const float* __restrict__ Emb, const _Float16* __restrict__ arena,
    Biases b, float* __restrict__ outFeat, float* __restrict__ outNbr,
    float* __restrict__ outClu) {
  extern __shared__ _Float16 smem[];
  _Float16* bufA = smem;            // 32 x 1024 f16 (64 KB)
  _Float16* bufB = smem + 32 * AW;  // 32 x 512  f16 (32 KB)
  const int lane = threadIdx.x;
  const size_t rowBase = (size_t)blockIdx.x * 32;

  // stage z[32,64] -> bufA (f16)
  for (int t = lane; t < 32 * 64; t += 32) {
    int r = t >> 6, c = t & 63;
    bufA[r * AW + c] = (_Float16)z[(rowBase + r) * 64 + c];
  }
  lds_fence();

  // trunk: z -> z1 -> z2  (z2[:,0:256]=feat_h, [256:512]=nb_h, [512:768]=cl_h)
  wave_gemm<0, 4>(arena + T_WLD, b.ld, bufA, AW, bufB, BW, nullptr, 0, nullptr, 0, 64, 8, 128, lane);
  wave_gemm<0, 4>(arena + T_WRD, b.rd, bufB, BW, bufA, AW, nullptr, 0, nullptr, 0, 128, 48, 768, lane);

  // feature head: sigmoid(feat_h @ Wfd + bfd) -> global
  wave_gemm<1, 4>(arena + T_WFD, b.fd, bufA, AW, nullptr, 0,
                  outFeat + rowBase * 512, 512, nullptr, 0, 256, 32, 512, lane);

  // cluster head
  wave_gemm<0, 4>(arena + T_WC1, b.c1, bufA + 512, AW, bufB, BW, nullptr, 0, nullptr, 0, 256, 16, 256, lane);
  wave_gemm<0, 4>(arena + T_WC2, b.c2, bufB, BW, bufB + 256, BW, nullptr, 0, nullptr, 0, 256, 16, 256, lane);
  wave_gemm<0, 4>(arena + T_WC3, b.c3, bufB + 256, BW, bufB, BW, nullptr, 0, nullptr, 0, 256, 16, 256, lane);
  wave_gemm<2, 4>(arena + T_WCD, b.cd, bufB, BW, nullptr, 0,
                  outClu + rowBase * 27, 27, nullptr, 0, 256, 4, 27, lane);
  lds_fence();

  // embedding gather: Emb[cluster_ind] -> bufB[:,0:128] (f16)
  for (int t = lane; t < 32 * 128; t += 32) {
    int r = t >> 7, c = t & 127;
    int ci = cind[rowBase + r];
    bufB[r * BW + c] = (_Float16)Emb[(size_t)ci * 128 + c];
  }
  lds_fence();

  // neighbor head: s = nb_h + relu(emb@Wcn + bcn) fused in epilogue
  wave_gemm<3, 4>(arena + T_WCN, b.cn, bufB, BW, bufA, AW, nullptr, 0,
                  bufA + 256, AW, 128, 16, 256, lane);
  wave_gemm<0, 4>(arena + T_WND, b.nd, bufA, AW, bufB, BW, nullptr, 0, nullptr, 0, 256, 32, 512, lane);
  wave_gemm<0, 4>(arena + T_WN1, b.n1, bufB, BW, bufA, AW, nullptr, 0, nullptr, 0, 512, 32, 512, lane);
  wave_gemm<0, 4>(arena + T_WN2, b.n2, bufA, AW, bufB, BW, nullptr, 0, nullptr, 0, 512, 32, 512, lane);
  wave_gemm<0, 4>(arena + T_WA0, b.a0, bufB, BW, bufA, AW, nullptr, 0, nullptr, 0, 512, 64, 1024, lane);
  wave_gemm<0, 4>(arena + T_WA1, b.a1, bufA, AW, bufB, BW, nullptr, 0, nullptr, 0, 1024, 32, 512, lane);
  wave_gemm<2, 4>(arena + T_WN3, b.n3, bufB, BW, nullptr, 0,
                  outNbr + rowBase * 101, 101, nullptr, 0, 512, 8, 101, lane);
}

extern "C" void kernel_launch(void* const* d_in, const int* in_sizes, int n_in,
                              void* d_out, int out_size, void* d_ws, size_t ws_size,
                              hipStream_t stream) {
  (void)n_in; (void)out_size; (void)ws_size;
  const float* z    = (const float*)d_in[0];
  const int*   cind = (const int*)d_in[1];
  const float* Emb  = (const float*)d_in[2];
  _Float16* arena = (_Float16*)d_ws;

  struct M { int idx; size_t off; int K, N; };
  const M mats[14] = {
      {3,  T_WLD, 64, 128},  {5,  T_WRD, 128, 768}, {7,  T_WFD, 256, 512},
      {9,  T_WC1, 256, 256}, {11, T_WC2, 256, 256}, {13, T_WC3, 256, 256},
      {15, T_WCD, 256, 27},  {17, T_WCN, 128, 256}, {19, T_WND, 256, 512},
      {21, T_WN1, 512, 512}, {23, T_WN2, 512, 512}, {25, T_WA0, 512, 1024},
      {27, T_WA1, 1024, 512},{29, T_WN3, 512, 101}};
  for (int i = 0; i < 14; ++i) {
    int KT = mats[i].K >> 5;
    int NT = (mats[i].N + 15) >> 4;
    int NTpad = (NT + 3) & ~3;           // multiple of NTB=4, zero-filled
    int total = KT * NTpad * 512;
    pack_weights<<<(total + 255) / 256, 256, 0, stream>>>(
        (const float*)d_in[mats[i].idx], arena + mats[i].off,
        mats[i].K, mats[i].N, KT, NTpad);
  }

  Biases bz{(const float*)d_in[4],  (const float*)d_in[6],  (const float*)d_in[8],
            (const float*)d_in[10], (const float*)d_in[12], (const float*)d_in[14],
            (const float*)d_in[16], (const float*)d_in[18], (const float*)d_in[20],
            (const float*)d_in[22], (const float*)d_in[24], (const float*)d_in[26],
            (const float*)d_in[28], (const float*)d_in[30]};

  const int Nn = in_sizes[1];  // number of nodes (cluster_ind length) = 131072
  float* out = (float*)d_out;
  float* outFeat = out;
  float* outNbr  = out + (size_t)Nn * 512;
  float* outClu  = out + (size_t)Nn * 613;

  fused_decoder<<<Nn / 32, 32, SMEM_BYTES, stream>>>(z, cind, Emb, arena, bz,
                                                     outFeat, outNbr, outClu);
}
